// MHConvAttention_6536940225038
// MI455X (gfx1250) — matrix operand order repaired and checked by reading, not compiled, not run on hardware
//
#include <hip/hip_runtime.h>

// ---------------------------------------------------------------------------
// MHConvAttention for MI455X (gfx1250): WMMA GEMMs (f16 in / f32 acc) with
// 4 N-tiles per wave (batched B-fragment loads for ILP), TDM
// (tensor_load_to_lds) staging for the lambda reduction, LDS-staged B panels,
// VALU kernels for depthwise convs / softmax / ECA.
// ---------------------------------------------------------------------------

typedef _Float16 h16 __attribute__((ext_vector_type(16)));
typedef _Float16 h8  __attribute__((ext_vector_type(8)));
typedef float    f8  __attribute__((ext_vector_type(8)));
typedef unsigned int  u32x4 __attribute__((ext_vector_type(4)));
typedef int           i32x4 __attribute__((ext_vector_type(4)));
typedef int           i32x8 __attribute__((ext_vector_type(8)));

#define BB   8
#define CC   128
#define HH   64
#define WW   64
#define NHD  4
#define HD   32
#define OUTC 2048
#define HWN  4096   // H*W
#define QKVC 384    // 3*C

#if defined(__AMDGCN__) && __has_builtin(__builtin_amdgcn_tensor_load_to_lds) && \
    __has_builtin(__builtin_amdgcn_s_wait_tensorcnt)
#define USE_TDM 1
#else
#define USE_TDM 0
#endif

#if USE_TDM
// 2D TDM load: tile 128 x 32 (f16) from a 4096-wide row-major tensor into LDS.
// clang-23 / therock form: 6 args (g0 v4u, g1 v8i, g2 v4i, g3 v4i, v8i, cpol).
__device__ __forceinline__ void tdm_load_k128x32(const _Float16* gptr, unsigned lds_off)
{
    unsigned long long ga = (unsigned long long)(uintptr_t)gptr;
    u32x4 g0;
    g0.x = 1u;                                        // count=1 (valid), no gather
    g0.y = lds_off;                                   // LDS byte address
    g0.z = (unsigned)(ga & 0xffffffffu);              // global_addr[31:0]
    g0.w = (unsigned)((ga >> 32) & 0x01ffffffu)       // global_addr[56:32]
         | (2u << 30);                                // type=2 (image)
    i32x8 g1;
    g1[0] = 0x10000;                                  // data_size=1 (2 bytes)
    g1[1] = (int)(4096u << 16);                       // tensor_dim0 = 4096 (lo16)
    g1[2] = (int)(32u << 16);                         // dim0 hi16=0 | tensor_dim1=32
    g1[3] = (int)(128u << 16);                        // dim1 hi16=0 | tile_dim0=128
    g1[4] = 32;                                       // tile_dim1=32, tile_dim2=0
    g1[5] = 4096;                                     // tensor_dim0_stride lo32
    g1[6] = 0;                                        // stride hi16 | dim1_stride lo16
    g1[7] = 0;
    i32x4 z4 = {0, 0, 0, 0};
    i32x8 z8 = {0, 0, 0, 0, 0, 0, 0, 0};
    __builtin_amdgcn_tensor_load_to_lds(g0, g1, z4, z4, z8, 0);
}
#endif

// ---------------------------------------------------------------------------
// K0: weight conversion + BN folding
// ---------------------------------------------------------------------------
__global__ __launch_bounds__(256) void prep_kernel(
    const float* __restrict__ qkv_w, const float* __restrict__ out_w,
    const float* __restrict__ g, const float* __restrict__ be,
    const float* __restrict__ mu, const float* __restrict__ var,
    _Float16* __restrict__ qkvw_h, _Float16* __restrict__ outw_h,
    float* __restrict__ bnscale, float* __restrict__ bnshift)
{
    int i = blockIdx.x * 256 + threadIdx.x;        // up to 524288
    if (i < QKVC * CC) qkvw_h[i] = (_Float16)qkv_w[i];
    if (i < OUTC * 2 * CC) outw_h[i] = (_Float16)out_w[i];
    if (i < OUTC) {
        float inv = g[i] * rsqrtf(var[i] + 1e-5f);
        bnscale[i] = inv;
        bnshift[i] = be[i] - mu[i] * inv;
    }
}

// ---------------------------------------------------------------------------
// K1: CPE depthwise 3x3 conv + residual -> x (f16)
// ---------------------------------------------------------------------------
__global__ __launch_bounds__(256) void cpe_kernel(
    const float* __restrict__ src, const float* __restrict__ cpe_w,
    _Float16* __restrict__ x_h)
{
    int idx = blockIdx.x * 256 + threadIdx.x;      // B*C*HW
    int p = idx & (HWN - 1);
    int c = (idx >> 12) & (CC - 1);
    int y = p >> 6, x = p & 63;
    const float* wp = cpe_w + c * 9;
    const float* s  = src + ((size_t)(idx >> 12)) * HWN;
    float acc = s[p];                               // residual
    #pragma unroll
    for (int dy = -1; dy <= 1; ++dy) {
        int yy = y + dy;
        if ((unsigned)yy < (unsigned)HH) {
            #pragma unroll
            for (int dx = -1; dx <= 1; ++dx) {
                int xx = x + dx;
                if ((unsigned)xx < (unsigned)WW)
                    acc += wp[(dy + 1) * 3 + (dx + 1)] * s[yy * WW + xx];
            }
        }
    }
    x_h[idx] = (_Float16)acc;
}

// ---------------------------------------------------------------------------
// K2: global average pool per (b,c)   (1024 blocks)
// ---------------------------------------------------------------------------
__global__ __launch_bounds__(256) void pool_kernel(
    const float* __restrict__ src, float* __restrict__ pooled)
{
    __shared__ float red[256];
    const float* s = src + (size_t)blockIdx.x * HWN;
    float a = 0.f;
    for (int i = threadIdx.x; i < HWN; i += 256) a += s[i];
    red[threadIdx.x] = a;
    __syncthreads();
    for (int st = 128; st > 0; st >>= 1) {
        if (threadIdx.x < st) red[threadIdx.x] += red[threadIdx.x + st];
        __syncthreads();
    }
    if (threadIdx.x == 0) pooled[blockIdx.x] = red[0] * (1.0f / HWN);
}

// ---------------------------------------------------------------------------
// K3: ECA conv1d (k=3, pad 1) + sigmoid
// ---------------------------------------------------------------------------
__global__ __launch_bounds__(256) void eca_kernel(
    const float* __restrict__ pooled, const float* __restrict__ w,
    float* __restrict__ attn)
{
    int i = blockIdx.x * 256 + threadIdx.x;
    if (i >= BB * CC) return;
    int c = i & (CC - 1);
    const float* pb = pooled + (i - c);
    float y = w[1] * pb[c];
    if (c > 0)      y += w[0] * pb[c - 1];
    if (c < CC - 1) y += w[2] * pb[c + 1];
    attn[i] = 1.f / (1.f + __expf(-y));
}

// ---------------------------------------------------------------------------
// WMMA GEMM:  Out[m,n] = sum_k A[m,k] * B[k,n]  per batch.
//   A: [Mtot, KTOT] f16 row-major (weights)
//   B: [KTOT, 4096] f16 per batch (batch stride KTOT*4096)
// Block tile = 128 M x 64 N; 8 waves stacked along M; each wave owns 16 rows
// and 4 N-subtiles. Per k-step: 1 A-fragment load, then ALL 4 B-fragment
// ds_loads issued before the 4 WMMAs (partial-counter waits instead of
// drain-per-WMMA). B panel in LDS transposed [n][k] for contiguous b128 reads.
//   FINAL=false: write f16 (qkv);  FINAL=true: f32 with BN + ReLU.
// ---------------------------------------------------------------------------
template <int KTOT, bool FINAL>
__global__ __launch_bounds__(256) void wmma_gemm_kernel(
    const _Float16* __restrict__ A, const _Float16* __restrict__ Bsrc,
    void* __restrict__ Out, int Mtot,
    const float* __restrict__ bnscale, const float* __restrict__ bnshift)
{
    constexpr int LDK = KTOT + 8;                    // pad to dodge bank conflicts
    __shared__ _Float16 Blds[64 * LDK];

    const int b       = blockIdx.z;
    const int colBase = blockIdx.x * 64;
    const int rowBlk  = blockIdx.y * 128;
    const _Float16* Bb = Bsrc + (size_t)b * KTOT * HWN;

    // cooperative transpose-load of the 64-column B panel
    const int t  = threadIdx.x;
    const int nn = t & 63;
    const int kc = t >> 6;                           // 0..3
    constexpr int KPT = KTOT / 4;
    #pragma unroll
    for (int i = 0; i < KPT; ++i) {
        int k = kc * KPT + i;
        Blds[nn * LDK + k] = Bb[(size_t)k * HWN + colBase + nn];
    }

    const int lane = t & 31;
    const int wave = t >> 5;
    const int m16  = lane & 15;
    const int half = lane >> 4;
    const int rowBase = rowBlk + wave * 16;

    const _Float16* Arow = A + (size_t)(rowBase + m16) * KTOT;
    __builtin_prefetch(Arow, 0, 0);                  // global_prefetch_b8
    __syncthreads();

    f8 acc0 = {}, acc1 = {}, acc2 = {}, acc3 = {};
    #pragma unroll
    for (int ks = 0; ks < KTOT / 32; ++ks) {
        const int k0 = ks * 32;
        // A fragment (16x32 f16 layout): lane m=lane&15, khalf=lane>>4
        h8 alo = *(const h8*)(Arow + k0 + 8 * half);
        h8 ahi = *(const h8*)(Arow + k0 + 16 + 8 * half);
        h16 afrag = __builtin_shufflevector(alo, ahi,
            0, 1, 2, 3, 4, 5, 6, 7, 8, 9, 10, 11, 12, 13, 14, 15);
        // batch all 4 B-fragment loads (independent ds_loads), then 4 WMMAs
        h16 bfrag[4];
        #pragma unroll
        for (int nt = 0; nt < 4; ++nt) {
            const _Float16* Bp = &Blds[(nt * 16 + m16) * LDK + k0 + 8 * half];
            h8 blo = *(const h8*)(Bp);
            h8 bhi = *(const h8*)(Bp + 16);
            bfrag[nt] = __builtin_shufflevector(blo, bhi,
                0, 1, 2, 3, 4, 5, 6, 7, 8, 9, 10, 11, 12, 13, 14, 15);
        }
        acc0 = __builtin_amdgcn_wmma_f32_16x16x32_f16(
            false, afrag, false, bfrag[0], (short)0, acc0, false, false);
        acc1 = __builtin_amdgcn_wmma_f32_16x16x32_f16(
            false, afrag, false, bfrag[1], (short)0, acc1, false, false);
        acc2 = __builtin_amdgcn_wmma_f32_16x16x32_f16(
            false, afrag, false, bfrag[2], (short)0, acc2, false, false);
        acc3 = __builtin_amdgcn_wmma_f32_16x16x32_f16(
            false, afrag, false, bfrag[3], (short)0, acc3, false, false);
    }

    // C/D layout: VGPR r -> (M = r + 8*half, N = lane&15)
    #pragma unroll
    for (int nt = 0; nt < 4; ++nt) {
        const f8& acc = (nt == 0) ? acc0 : (nt == 1) ? acc1 : (nt == 2) ? acc2 : acc3;
        const int col = colBase + nt * 16 + m16;
        if (!FINAL) {
            _Float16* Ob = (_Float16*)Out + (size_t)b * Mtot * HWN;
            #pragma unroll
            for (int r = 0; r < 8; ++r) {
                int row = rowBase + r + 8 * half;
                Ob[(size_t)row * HWN + col] = (_Float16)acc[r];
            }
        } else {
            float* Ob = (float*)Out + (size_t)b * Mtot * HWN;
            #pragma unroll
            for (int r = 0; r < 8; ++r) {
                int row = rowBase + r + 8 * half;
                float v = acc[r] * bnscale[row] + bnshift[row];
                Ob[(size_t)row * HWN + col] = fmaxf(v, 0.f);
            }
        }
    }
}

// ---------------------------------------------------------------------------
// K5: per-row softmax stats for k  (row = b*128 + c, over 4096 positions)
// ---------------------------------------------------------------------------
__global__ __launch_bounds__(256) void softstat_kernel(
    const _Float16* __restrict__ qkv_h,
    float* __restrict__ rowmax, float* __restrict__ rowsum)
{
    __shared__ float red[256];
    const int row = blockIdx.x;                       // 0..1023
    const int b = row >> 7, c = row & 127;
    const _Float16* kr = qkv_h + ((size_t)b * QKVC + CC + c) * HWN;
    float m = -3.0e38f;
    for (int i = threadIdx.x; i < HWN; i += 256) m = fmaxf(m, (float)kr[i]);
    red[threadIdx.x] = m;
    __syncthreads();
    for (int st = 128; st > 0; st >>= 1) {
        if (threadIdx.x < st)
            red[threadIdx.x] = fmaxf(red[threadIdx.x], red[threadIdx.x + st]);
        __syncthreads();
    }
    const float mg = red[0];
    __syncthreads();
    float s = 0.f;
    for (int i = threadIdx.x; i < HWN; i += 256) s += __expf((float)kr[i] - mg);
    red[threadIdx.x] = s;
    __syncthreads();
    for (int st = 128; st > 0; st >>= 1) {
        if (threadIdx.x < st) red[threadIdx.x] += red[threadIdx.x + st];
        __syncthreads();
    }
    if (threadIdx.x == 0) { rowmax[row] = mg; rowsum[row] = red[0]; }
}

// ---------------------------------------------------------------------------
// K6: content lambda  lam[bh][i][o] = scale/sum_i * sum_n exp(k[i][n]-max_i)*v[o][n]
// one block per (b,h); k/v tiles staged into LDS by the Tensor Data Mover
// (tensor_load_to_lds + s_wait_tensorcnt), then exp/convert pass, then MAC.
// ---------------------------------------------------------------------------
__global__ __launch_bounds__(256) void lambda_kernel(
    const _Float16* __restrict__ qkv_h,
    const float* __restrict__ rowmax, const float* __restrict__ rowsum,
    float* __restrict__ lam)
{
    __shared__ _Float16 kraw[HD * 128];
    __shared__ _Float16 vraw[HD * 128];
    __shared__ float ke[HD * 128];
    __shared__ float vv[HD * 128];
    const int bh = blockIdx.x;                        // 0..31
    const int b = bh >> 2, h = bh & 3;
    const int t = threadIdx.x;
    const _Float16* kbase = qkv_h + ((size_t)b * QKVC + CC + h * HD) * HWN;
    const _Float16* vbase = qkv_h + ((size_t)b * QKVC + 2 * CC + h * HD) * HWN;

    const int i0 = (t * 4) >> 5;                      // same i for all 4 pairs
    const int o0 = (t * 4) & 31;
    float acc0 = 0.f, acc1 = 0.f, acc2 = 0.f, acc3 = 0.f;

    const int lr = t >> 3;                            // row 0..31 for staging
    const int ln = (t & 7) * 16;                      // 16 consecutive positions
    const float km = rowmax[b * CC + h * HD + lr];

    for (int ch = 0; ch < HWN / 128; ++ch) {
        const int nb = ch * 128;
#if USE_TDM
        if (t == 0) {
            tdm_load_k128x32(kbase + nb, (unsigned)(uintptr_t)(void*)kraw);
            tdm_load_k128x32(vbase + nb, (unsigned)(uintptr_t)(void*)vraw);
            __builtin_amdgcn_s_wait_tensorcnt(0);
        }
        __syncthreads();
#else
        #pragma unroll
        for (int e = 0; e < 16; ++e) {
            int n = ln + e;
            kraw[lr * 128 + n] = kbase[(size_t)lr * HWN + nb + n];
            vraw[lr * 128 + n] = vbase[(size_t)lr * HWN + nb + n];
        }
        __syncthreads();
#endif
        #pragma unroll
        for (int e = 0; e < 16; ++e) {
            int n = ln + e;
            ke[lr * 128 + n] = __expf((float)kraw[lr * 128 + n] - km);
            vv[lr * 128 + n] = (float)vraw[lr * 128 + n];
        }
        __syncthreads();
        #pragma unroll 8
        for (int n = 0; n < 128; ++n) {
            float kv = ke[i0 * 128 + n];
            acc0 += kv * vv[(o0 + 0) * 128 + n];
            acc1 += kv * vv[(o0 + 1) * 128 + n];
            acc2 += kv * vv[(o0 + 2) * 128 + n];
            acc3 += kv * vv[(o0 + 3) * 128 + n];
        }
        __syncthreads();
    }
    const float sc = 0.17677669529663687f /* 1/sqrt(32) */
                     / rowsum[b * CC + h * HD + i0];
    float* lp = lam + (size_t)bh * HD * HD + i0 * HD + o0;
    lp[0] = acc0 * sc; lp[1] = acc1 * sc; lp[2] = acc2 * sc; lp[3] = acc3 * sc;
}

// ---------------------------------------------------------------------------
// K7: result1 = content_out + q * pos_lambda  -> cat[:, 0:128, :]
// thread = one (bh, pixel); lambda + rel_pos in LDS
// ---------------------------------------------------------------------------
__global__ __launch_bounds__(256) void fuse1_kernel(
    const _Float16* __restrict__ qkv_h, const float* __restrict__ lam,
    const float* __restrict__ rel_pos, _Float16* __restrict__ cat_h)
{
    __shared__ float slam[HD * HD];
    __shared__ float srp[HD * 25];
    const int idx = blockIdx.x * 256 + threadIdx.x;   // 0..131071
    const int p  = idx & (HWN - 1);
    const int bh = idx >> 12;
    const int b = bh >> 2, h = bh & 3;
    for (int i = threadIdx.x; i < HD * HD; i += 256) slam[i] = lam[(size_t)bh * HD * HD + i];
    for (int i = threadIdx.x; i < HD * 25; i += 256) srp[i] = rel_pos[i];
    __syncthreads();

    const _Float16* qb = qkv_h + ((size_t)b * QKVC + h * HD) * HWN;
    const _Float16* vb = qkv_h + ((size_t)b * QKVC + 2 * CC + h * HD) * HWN;
    float qreg[HD];
    #pragma unroll
    for (int i = 0; i < HD; ++i) qreg[i] = (float)qb[(size_t)i * HWN + p];

    const int py = p >> 6, px = p & 63;
    _Float16* cb = cat_h + ((size_t)b * (2 * CC) + h * HD) * HWN;

    #pragma unroll 1
    for (int o = 0; o < HD; ++o) {
        float content = 0.f;
        #pragma unroll
        for (int i = 0; i < HD; ++i) content += qreg[i] * slam[i * HD + o];
        float pos = 0.f;
        const _Float16* vo = vb + (size_t)o * HWN;
        #pragma unroll
        for (int dy = -2; dy <= 2; ++dy) {
            int yy = py + dy;
            if ((unsigned)yy < (unsigned)HH) {
                #pragma unroll
                for (int dx = -2; dx <= 2; ++dx) {
                    int xx = px + dx;
                    if ((unsigned)xx < (unsigned)WW)
                        pos += srp[o * 25 + (dy + 2) * 5 + (dx + 2)]
                             * (float)vo[yy * WW + xx];
                }
            }
        }
        cb[(size_t)o * HWN + p] = (_Float16)(content + qreg[o] * pos);
    }
}

// ---------------------------------------------------------------------------
// K8: result2 = src * attn -> cat[:, 128:256, :]
// ---------------------------------------------------------------------------
__global__ __launch_bounds__(256) void fuse2_kernel(
    const float* __restrict__ src, const float* __restrict__ attn,
    _Float16* __restrict__ cat_h)
{
    int idx = blockIdx.x * 256 + threadIdx.x;         // B*C*HW
    int p  = idx & (HWN - 1);
    int bc = idx >> 12;
    int b = bc >> 7, c = bc & 127;
    float v = src[idx] * attn[bc];
    cat_h[((size_t)b * (2 * CC) + CC + c) * HWN + p] = (_Float16)v;
}

// ---------------------------------------------------------------------------
// launch
// ---------------------------------------------------------------------------
extern "C" void kernel_launch(void* const* d_in, const int* in_sizes, int n_in,
                              void* d_out, int out_size, void* d_ws, size_t ws_size,
                              hipStream_t stream) {
    (void)in_sizes; (void)n_in; (void)out_size; (void)ws_size;
    const float* src     = (const float*)d_in[0];
    const float* cpe_w   = (const float*)d_in[1];
    const float* qkv_w   = (const float*)d_in[2];
    const float* rel_pos = (const float*)d_in[3];
    const float* c1d_w   = (const float*)d_in[4];
    const float* out_w   = (const float*)d_in[5];
    const float* bn_g    = (const float*)d_in[6];
    const float* bn_b    = (const float*)d_in[7];
    const float* bn_m    = (const float*)d_in[8];
    const float* bn_v    = (const float*)d_in[9];
    float* out = (float*)d_out;

    char* ws = (char*)d_ws;
    size_t off = 0;
    auto take = [&](size_t bytes) -> char* {
        char* p = ws + off;
        off = (off + bytes + 255) & ~(size_t)255;
        return p;
    };
    _Float16* qkvw_h  = (_Float16*)take((size_t)QKVC * CC * 2);
    _Float16* outw_h  = (_Float16*)take((size_t)OUTC * 2 * CC * 2);
    float*    bnscale = (float*)take(OUTC * 4);
    float*    bnshift = (float*)take(OUTC * 4);
    float*    pooled  = (float*)take(BB * CC * 4);
    float*    attn    = (float*)take(BB * CC * 4);
    float*    rowmax  = (float*)take(BB * CC * 4);
    float*    rowsum  = (float*)take(BB * CC * 4);
    float*    lam     = (float*)take((size_t)BB * NHD * HD * HD * 4);
    _Float16* x_h     = (_Float16*)take((size_t)BB * CC * HWN * 2);
    _Float16* qkv_h   = (_Float16*)take((size_t)BB * QKVC * HWN * 2);
    _Float16* cat_h   = (_Float16*)take((size_t)BB * 2 * CC * HWN * 2);

    prep_kernel<<<(OUTC * 2 * CC) / 256, 256, 0, stream>>>(
        qkv_w, out_w, bn_g, bn_b, bn_m, bn_v, qkvw_h, outw_h, bnscale, bnshift);

    cpe_kernel<<<(BB * CC * HWN) / 256, 256, 0, stream>>>(src, cpe_w, x_h);

    pool_kernel<<<BB * CC, 256, 0, stream>>>(src, pooled);
    eca_kernel<<<(BB * CC + 255) / 256, 256, 0, stream>>>(pooled, c1d_w, attn);

    // QKV GEMM: M=384, K=128, per-batch N=4096 (64 cols/block)
    wmma_gemm_kernel<CC, false><<<dim3(HWN / 64, QKVC / 128, BB), 256, 0, stream>>>(
        qkvw_h, x_h, (void*)qkv_h, QKVC, nullptr, nullptr);

    softstat_kernel<<<BB * CC, 256, 0, stream>>>(qkv_h, rowmax, rowsum);
    lambda_kernel<<<BB * NHD, 256, 0, stream>>>(qkv_h, rowmax, rowsum, lam);

    fuse1_kernel<<<(BB * NHD * HWN) / 256, 256, 0, stream>>>(qkv_h, lam, rel_pos, cat_h);
    fuse2_kernel<<<(BB * CC * HWN) / 256, 256, 0, stream>>>(src, attn, cat_h);

    // Output GEMM: M=2048, K=256, per-batch N=4096, fused BN + ReLU (64 cols/block)
    wmma_gemm_kernel<2 * CC, true><<<dim3(HWN / 64, OUTC / 128, BB), 256, 0, stream>>>(
        outw_h, cat_h, (void*)out, OUTC, bnscale, bnshift);
}
